// Transformer_87462714016302
// MI455X (gfx1250) — matrix operand order repaired and checked by reading, not compiled, not run on hardware
//
#include <hip/hip_runtime.h>

typedef _Float16 f16;
typedef __attribute__((ext_vector_type(16))) _Float16 v16h;
typedef __attribute__((ext_vector_type(8)))  _Float16 v8h;
typedef __attribute__((ext_vector_type(8)))  float    v8f;

#define DIM   192
#define NHEAD 6
#define HD    32
#define NTOK  64
#define QK_SCALE 0.17677669529663687f   // 32^-0.5

union U16h { v16h v; v8h h[2]; };
union U8f  { v8f v; float e[8]; };
union U8h  { v8h v; f16 e[8]; };

__device__ __forceinline__ v8f wmma_f16(v16h a, v16h b, v8f c) {
  // D = A(16x32 f16) * B(32x16 f16) + C(16x16 f32)
  return __builtin_amdgcn_wmma_f32_16x16x32_f16(
      /*neg_a=*/false, a, /*neg_b=*/false, b,
      /*c_mod=*/(short)0, c, /*reuse_a=*/false, /*reuse_b=*/false);
}

// A fragment: 16x32 f16 tile from row-major [M][K] buffer (stride in halves).
// Lane layout: row = lane%16, element e -> K = 16*(lane>=16) + e (contiguous).
__device__ __forceinline__ v16h load_a_frag(const f16* base, int stride,
                                            int m0, int k0, int lm, int hi) {
  const f16* p = base + (m0 + lm) * stride + k0 + hi * 16;
  U16h u;
  u.h[0] = *(const v8h*)p;
  u.h[1] = *(const v8h*)(p + 8);
  return u.v;
}

// B fragment: 32x16 f16 tile, B[k][n] taken from row-major [N][K] buffer
// (i.e. the "weight" orientation W[n][k]): lane col n = n0+lane%16,
// element e -> K = k0 + 16*(lane>=16) + e  (16 contiguous halves = 32B).
__device__ __forceinline__ v16h load_b_frag(const f16* base, int stride,
                                            int n0, int k0, int lm, int hi) {
  return *(const v16h*)(base + (n0 + lm) * stride + k0 + hi * 16);
}

__global__ void convert_weights(const float* __restrict__ qkv_w, int nq,
                                const float* __restrict__ proj_w, int np,
                                f16* __restrict__ wq, f16* __restrict__ wp) {
  int i = blockIdx.x * 256 + threadIdx.x;
  if (i < nq) wq[i] = (f16)qkv_w[i];
  if (i < np) wp[i] = (f16)proj_w[i];
}

__global__ void __launch_bounds__(256)
swin_window_attn(const float* __restrict__ x,
                 const f16* __restrict__ wq,      // 576x192 f16
                 const float* __restrict__ qkv_b, // 576
                 const f16* __restrict__ wp,      // 192x192 f16
                 const float* __restrict__ proj_b,// 192
                 const float* __restrict__ relt,  // 225x6
                 float* __restrict__ out) {
  extern __shared__ char smem[];
  f16*   qs = (f16*)smem;            // [64][192] scaled Q            24576 B
  f16*   ks = qs + NTOK * DIM;       // [64][192] K                   24576 B
  f16*   vt = ks + NTOK * DIM;       // [192][64] V transposed        24576 B
  f16*   xs = vt + DIM * NTOK;       // [64][192] x (f16), reused as O 24576 B
  f16*   ps = xs + NTOK * DIM;       // [6][64][64] attention probs   49152 B
  float* bl = (float*)(ps + NHEAD * NTOK * NTOK); // [225][6] bias     5400 B

  const int tid  = threadIdx.x;
  const int lane = tid & 31;
  // Wave id is uniform across the wave: pin it to an SGPR so all per-wave
  // tile math (mt / ntb / f0 / which) is scalar -> s_cmp/s_cbranch instead
  // of v_cmpx exec juggling and VALU magic-number division.
  const int wv   = __builtin_amdgcn_readfirstlane(tid >> 5);   // 0..7
  const int lm   = lane & 15;
  const int hi   = lane >> 4;
  const int b    = blockIdx.x;

  // ---------------- Phase 1: stage inputs ----------------
  const float* xb = x + (size_t)b * (NTOK * DIM);
  for (int i = tid; i < NTOK * DIM; i += 256) xs[i] = (f16)xb[i];
  for (int i = tid; i < 225 * NHEAD; i += 256) bl[i] = relt[i];
  __syncthreads();

  // ---------------- Phase 2: QKV = x @ W^T + b ----------------
  {
    const int mt  = wv & 3;          // token row-tile (fixed per wave)
    const int ntb = (wv >> 2) * 18;  // 36 feature tiles split over 2 halves
    v16h A[6];
    #pragma unroll
    for (int kk = 0; kk < 6; ++kk)
      A[kk] = load_a_frag(xs, DIM, mt * 16, kk * 32, lm, hi);
    for (int t = 0; t < 18; ++t) {
      const int f0 = (ntb + t) * 16;                 // wave-uniform (scalar)
      U8f acc; acc.v = (v8f){};
      #pragma unroll
      for (int kk = 0; kk < 6; ++kk)
        acc.v = wmma_f16(A[kk], load_b_frag(wq, DIM, f0, kk * 32, lm, hi), acc.v);
      const float bias = qkv_b[f0 + lm];
      if (f0 < 192) {                                // Q (scaled)
        const int c = f0 + lm;
        #pragma unroll
        for (int r = 0; r < 8; ++r)
          qs[(mt * 16 + hi * 8 + r) * DIM + c] = (f16)((acc.e[r] + bias) * QK_SCALE);
      } else if (f0 < 384) {                         // K
        const int c = f0 - 192 + lm;
        #pragma unroll
        for (int r = 0; r < 8; ++r)
          ks[(mt * 16 + hi * 8 + r) * DIM + c] = (f16)(acc.e[r] + bias);
      } else {                                       // V transposed: vt[c][m]
        const int c = f0 - 384 + lm;
        U8h h;
        #pragma unroll
        for (int r = 0; r < 8; ++r) h.e[r] = (f16)(acc.e[r] + bias);
        *(v8h*)(vt + c * NTOK + mt * 16 + hi * 8) = h.v;
      }
    }
  }
  __syncthreads();

  // ---------------- Phase 3: per-head attention (wave = head) ----------------
  if (wv < NHEAD) {                                  // scalar branch
    const int h  = wv;
    f16*      ph = ps + h * NTOK * NTOK;
    v16h Bk[4];                     // K^T fragments: B[k][n] = K[token n][dim k]
    #pragma unroll
    for (int nt = 0; nt < 4; ++nt)
      Bk[nt] = load_b_frag(ks, DIM, nt * 16, h * HD, lm, hi);

    for (int mt = 0; mt < 4; ++mt) {
      v16h Aq = load_a_frag(qs, DIM, mt * 16, h * HD, lm, hi);
      U8f S[4];
      #pragma unroll
      for (int nt = 0; nt < 4; ++nt) {
        v8f z = (v8f){};
        S[nt].v = wmma_f16(Aq, Bk[nt], z);     // head_dim 32 == one WMMA
      }
      #pragma unroll
      for (int r = 0; r < 8; ++r) {
        const int m = mt * 16 + hi * 8 + r;
        const int ihh = m >> 3, iww = m & 7;
        float mx = -1e30f;
        #pragma unroll
        for (int nt = 0; nt < 4; ++nt) {
          const int n   = nt * 16 + lm;
          const int idx = (ihh - (n >> 3) + 7) * 15 + (iww - (n & 7) + 7);
          S[nt].e[r] += bl[idx * NHEAD + h];
          mx = fmaxf(mx, S[nt].e[r]);
        }
        #pragma unroll
        for (int off = 1; off < 16; off <<= 1)   // reduce over N (16-lane group)
          mx = fmaxf(mx, __shfl_xor(mx, off, 32));
        float sm = 0.f;
        #pragma unroll
        for (int nt = 0; nt < 4; ++nt) {
          const float e = __expf(S[nt].e[r] - mx);
          S[nt].e[r] = e; sm += e;
        }
        #pragma unroll
        for (int off = 1; off < 16; off <<= 1)
          sm += __shfl_xor(sm, off, 32);
        const float inv = 1.f / sm;
        #pragma unroll
        for (int nt = 0; nt < 4; ++nt)
          ph[m * NTOK + nt * 16 + lm] = (f16)(S[nt].e[r] * inv);
      }
    }
    // O = P @ V   (A = P rows from LDS, B = V via transposed vt -> contiguous)
    for (int mt = 0; mt < 4; ++mt) {
      v16h Ap0 = load_a_frag(ph, NTOK, mt * 16, 0,  lm, hi);
      v16h Ap1 = load_a_frag(ph, NTOK, mt * 16, 32, lm, hi);
      #pragma unroll
      for (int nt2 = 0; nt2 < 2; ++nt2) {
        U8f o; o.v = (v8f){};
        o.v = wmma_f16(Ap0, load_b_frag(vt, NTOK, h * HD + nt2 * 16, 0,  lm, hi), o.v);
        o.v = wmma_f16(Ap1, load_b_frag(vt, NTOK, h * HD + nt2 * 16, 32, lm, hi), o.v);
        #pragma unroll
        for (int r = 0; r < 8; ++r)
          xs[(mt * 16 + hi * 8 + r) * DIM + h * HD + nt2 * 16 + lm] = (f16)o.e[r];
      }
    }
  }
  __syncthreads();

  // ---------------- Phase 4: out = O @ proj_w^T + proj_b ----------------
  {
    const int mt  = wv & 3;
    const int ntb = (wv >> 2) * 6;   // 12 feature tiles over 2 halves
    v16h A[6];
    #pragma unroll
    for (int kk = 0; kk < 6; ++kk)
      A[kk] = load_a_frag(xs, DIM, mt * 16, kk * 32, lm, hi);
    float* ob = out + (size_t)b * (NTOK * DIM);
    for (int t = 0; t < 6; ++t) {
      const int f0 = (ntb + t) * 16;
      U8f acc; acc.v = (v8f){};
      #pragma unroll
      for (int kk = 0; kk < 6; ++kk)
        acc.v = wmma_f16(A[kk], load_b_frag(wp, DIM, f0, kk * 32, lm, hi), acc.v);
      const float pb = proj_b[f0 + lm];
      #pragma unroll
      for (int r = 0; r < 8; ++r)
        ob[(mt * 16 + hi * 8 + r) * DIM + f0 + lm] = acc.e[r] + pb;
    }
  }
}

extern "C" void kernel_launch(void* const* d_in, const int* in_sizes, int n_in,
                              void* d_out, int out_size, void* d_ws, size_t ws_size,
                              hipStream_t stream) {
  const float* x      = (const float*)d_in[0];
  const float* qkv_w  = (const float*)d_in[1];
  const float* qkv_b  = (const float*)d_in[2];
  const float* proj_w = (const float*)d_in[3];
  const float* proj_b = (const float*)d_in[4];
  const float* relt   = (const float*)d_in[5];
  (void)in_sizes; (void)n_in; (void)out_size; (void)ws_size;

  f16* wq = (f16*)d_ws;              // 576*192 halves
  f16* wp = wq + 3 * DIM * DIM;      // 192*192 halves
  const int nq = 3 * DIM * DIM, np = DIM * DIM;
  convert_weights<<<(nq + 255) / 256, 256, 0, stream>>>(qkv_w, nq, proj_w, np, wq, wp);

  const int    nwin = 4096;
  const size_t smem = (size_t)(4 * NTOK * DIM + NHEAD * NTOK * NTOK) * sizeof(f16)
                    + (size_t)(225 * NHEAD) * sizeof(float);   // ~152.9 KB
  swin_window_attn<<<nwin, 256, smem, stream>>>(x, wq, qkv_b, wp, proj_b, relt,
                                                (float*)d_out);
}
